// Encoder_42150809043101
// MI455X (gfx1250) — matrix operand order repaired and checked by reading, not compiled, not run on hardware
//
#include <hip/hip_runtime.h>
#include <hip/hip_bf16.h>
#include <stdint.h>

#define SEQ    512
#define BATCH  32
#define HIDN   1024
#define FFDIM  4096
#define HEADS  8
#define NX     6
#define NTOK   (SEQ*BATCH)            // 16384 rows of the activation matrix
#define NACT   ((size_t)NTOK*HIDN)    // 16777216 elements
#define EPS    1e-5f

typedef __attribute__((ext_vector_type(16))) __bf16 bf16x16;
typedef __attribute__((ext_vector_type(8)))  __bf16 bf16x8;
typedef __attribute__((ext_vector_type(4)))  __bf16 bf16x4;
typedef __attribute__((ext_vector_type(8)))  float  f32x8;

typedef int v4i __attribute__((vector_size(16)));
typedef __attribute__((address_space(1))) v4i v4i_as1;   // global
typedef __attribute__((address_space(3))) v4i v4i_as3;   // LDS

#if __has_builtin(__builtin_amdgcn_global_load_async_to_lds_b128)
#define HAVE_ASYNC 1
#else
#define HAVE_ASYNC 0
#endif

// fp32 -> bf16 round-to-nearest-even via integer math
__device__ __forceinline__ __bf16 f2bf(float f) {
  unsigned u = __builtin_bit_cast(unsigned, f);
  unsigned r = u + 0x7FFFu + ((u >> 16) & 1u);
  unsigned short h = (unsigned short)(r >> 16);
  return __builtin_bit_cast(__bf16, h);
}

// 16B global -> 16B LDS copy; CDNA5 async (ASYNCcnt) when available
__device__ __forceinline__ void copy_b128_to_lds(const void* g, void* l) {
#if HAVE_ASYNC
  __builtin_amdgcn_global_load_async_to_lds_b128(
      (v4i_as1*)(uintptr_t)g, (v4i_as3*)(uintptr_t)l, 0, 0);
#else
  *reinterpret_cast<uint4*>(l) = *reinterpret_cast<const uint4*>(g);
#endif
}

__device__ __forceinline__ void wait_async_le2() {
#if HAVE_ASYNC
#if __has_builtin(__builtin_amdgcn_s_wait_asynccnt)
  __builtin_amdgcn_s_wait_asynccnt(2);
#else
  asm volatile("s_wait_asynccnt 0x2" ::: "memory");
#endif
#endif
}
__device__ __forceinline__ void wait_async_0() {
#if HAVE_ASYNC
#if __has_builtin(__builtin_amdgcn_s_wait_asynccnt)
  __builtin_amdgcn_s_wait_asynccnt(0);
#else
  asm volatile("s_wait_asynccnt 0x0" ::: "memory");
#endif
#endif
}

// Build a 16-element bf16 fragment from two 16B LDS chunks (2x ds_load_b128)
__device__ __forceinline__ bf16x16 frag16(const __bf16* p0, const __bf16* p1) {
  bf16x8 lo = *reinterpret_cast<const bf16x8*>(p0);
  bf16x8 hi = *reinterpret_cast<const bf16x8*>(p1);
  return __builtin_shufflevector(lo, hi, 0,1,2,3,4,5,6,7,8,9,10,11,12,13,14,15);
}

// ---------------------------------------------------------------------------
// One-time per launch: W[K,N] fp32 -> Wt[N,K] bf16 (transposed, WMMA-ready)
// 64x64 tiles through LDS so both sides are coalesced.
// ---------------------------------------------------------------------------
__global__ __launch_bounds__(256)
void wconv_kernel(const float* __restrict__ W, __bf16* __restrict__ Wt, int K, int N)
{
  __shared__ float tile[64][68];          // pad to dodge bank conflicts
  const int n0 = blockIdx.x * 64, k0 = blockIdx.y * 64;
  const int tid = threadIdx.x;
#pragma unroll
  for (int i = 0; i < 4; i++) {
    int lin = tid + i * 256;
    int r = lin >> 4;                     // k row 0..63
    int c = (lin & 15) << 2;              // n col
    float4 v = *reinterpret_cast<const float4*>(W + (size_t)(k0 + r) * N + n0 + c);
    tile[r][c] = v.x; tile[r][c + 1] = v.y; tile[r][c + 2] = v.z; tile[r][c + 3] = v.w;
  }
  __syncthreads();
#pragma unroll
  for (int i = 0; i < 4; i++) {
    int lin = tid + i * 256;
    int n = lin >> 4;                     // 0..63
    int k = (lin & 15) << 2;              // 0..60
    bf16x4 o = { f2bf(tile[k][n]), f2bf(tile[k + 1][n]),
                 f2bf(tile[k + 2][n]), f2bf(tile[k + 3][n]) };
    *reinterpret_cast<bf16x4*>(Wt + (size_t)(n0 + n) * K + k0 + k) = o;
  }
}

// ---------------------------------------------------------------------------
// GEMM: C[M,N] = A[M,K](fp32) @ Wt[N,K](bf16, pre-transposed) + bias, opt ReLU.
// Block tile 128x128, BK=32, double-buffered LDS.
// A: global fp32 -> regs (prefetch) -> bf16 LDS.   B: pure 16B async copies.
// 256 threads = 8 waves, each wave owns 32x64 (8 WMMA accumulators).
// ---------------------------------------------------------------------------
__global__ __launch_bounds__(256)
void gemm_bias_kernel(const float* __restrict__ A, const __bf16* __restrict__ Bt,
                      const float* __restrict__ bias, float* __restrict__ C,
                      int M, int Kdim, int N, int relu)
{
  __shared__ __bf16 As[2][128 * 32];      // [m][k], 8KB each
  __shared__ __bf16 Bs[2][128 * 32];      // [n][k], 8KB each

  const int tid  = threadIdx.x;
  const int lane = tid & 31, wid = tid >> 5;
  const int half = lane >> 4, l16 = lane & 15;
  const int wm = wid & 3;                 // 4 waves over M: 32 rows each
  const int wn = wid >> 2;                // 2 waves over N: 64 cols each
  const int m0 = blockIdx.x * 128, n0 = blockIdx.y * 128;
  (void)M;

  f32x8 acc[2][4] = {};

  // issue B-tile copy (bf16 [n][k] rows, 64B per row = 4 x 16B chunks)
  auto issueB = [&](int buf, int k0) {
#pragma unroll
    for (int i = 0; i < 2; i++) {
      int lin = tid + i * 256;
      int n = lin >> 2;                   // 0..127
      int c = (lin & 3) << 3;             // element offset 0,8,16,24
      const __bf16* g = Bt + (size_t)(n0 + n) * Kdim + k0 + c;
      copy_b128_to_lds(g, &Bs[buf][n * 32 + c]);
    }
  };
  auto prefetchA = [&](float4* pa, int k0) {
#pragma unroll
    for (int i = 0; i < 4; i++) {
      int lin = tid + i * 256;
      int row = lin >> 3;
      int col = (lin & 7) << 2;
      pa[i] = *reinterpret_cast<const float4*>(A + (size_t)(m0 + row) * Kdim + k0 + col);
    }
  };
  auto storeA = [&](const float4* pa, int buf) {
#pragma unroll
    for (int i = 0; i < 4; i++) {
      int lin = tid + i * 256;
      int row = lin >> 3;
      int col = (lin & 7) << 2;
      __bf16* d = &As[buf][row * 32 + col];
      d[0] = f2bf(pa[i].x); d[1] = f2bf(pa[i].y);
      d[2] = f2bf(pa[i].z); d[3] = f2bf(pa[i].w);
    }
  };

  // ---- prologue: fill buffer 0 ----
  {
    float4 pa[4];
    prefetchA(pa, 0);
    issueB(0, 0);
    storeA(pa, 0);
  }

  const int nk = Kdim >> 5;
  for (int it = 0; it < nk; ++it) {
    const int cur = it & 1, nxt = cur ^ 1;
    const bool haveNext = (it + 1 < nk);
    float4 pa[4];
    if (haveNext) {
      prefetchA(pa, (it + 1) << 5);
      issueB(nxt, (it + 1) << 5);
      wait_async_le2();                   // cur tile's async copies done
    } else {
      wait_async_0();
    }
    __syncthreads();                      // cur As/Bs visible to all waves
    if (haveNext) storeA(pa, nxt);        // fill other buffer while computing

    bf16x16 af[2], bfr[4];
#pragma unroll
    for (int tm = 0; tm < 2; tm++) {
      int m = wm * 32 + tm * 16 + l16;
      af[tm] = frag16(&As[cur][m * 32 + half * 8], &As[cur][m * 32 + 16 + half * 8]);
    }
#pragma unroll
    for (int tn = 0; tn < 4; tn++) {
      int n = wn * 64 + tn * 16 + l16;
      bfr[tn] = frag16(&Bs[cur][n * 32 + half * 8], &Bs[cur][n * 32 + 16 + half * 8]);
    }
#pragma unroll
    for (int tm = 0; tm < 2; tm++)
#pragma unroll
      for (int tn = 0; tn < 4; tn++)
        acc[tm][tn] = __builtin_amdgcn_wmma_f32_16x16x32_bf16(
            false, af[tm], false, bfr[tn], (short)0, acc[tm][tn], false, false);
    __syncthreads();                      // done reading cur before it+2 refills it
  }

  // ---- epilogue: bias (+ReLU), store per C/D layout ----
#pragma unroll
  for (int tm = 0; tm < 2; tm++) {
#pragma unroll
    for (int tn = 0; tn < 4; tn++) {
      int n = n0 + wn * 64 + tn * 16 + l16;
      float bv = bias[n];
#pragma unroll
      for (int r = 0; r < 8; r++) {
        int m = m0 + wm * 32 + tm * 16 + r + half * 8;
        float v = acc[tm][tn][r] + bv;
        if (relu) v = fmaxf(v, 0.f);
        C[(size_t)m * N + n] = v;
      }
    }
  }
}

// ---------------------------------------------------------------------------
// Attention per (b, head): E = Q_sub K_sub^T / sqrt(128); softmax rows;
// O = att @ V_sub.  Q_sub rows are sequence positions s = t*8+hd (t=0..63).
// Output scattered to R = t*256 + b*8 + hd (reference reshape scramble).
// ---------------------------------------------------------------------------
__global__ __launch_bounds__(256)
void attention_kernel(const float* __restrict__ Qg, const float* __restrict__ Kg,
                      const float* __restrict__ Vg, float* __restrict__ Og)
{
  __shared__ __bf16 Qs[64 * 32];
  __shared__ __bf16 Ks[64 * 32];
  __shared__ float  Es[64 * 64];
  __shared__ __bf16 Ab[64 * 64];
  __shared__ __bf16 Vt[128 * 64];

  const int b = blockIdx.x >> 3, hd = blockIdx.x & 7;
  const int tid  = threadIdx.x;
  const int lane = tid & 31, wid = tid >> 5;
  const int half = lane >> 4, l16 = lane & 15;

  // ---------------- phase 1: E = Q K^T ----------------
  const int tm1 = wid & 3;
  const int tn1 = (wid >> 2) * 2;
  f32x8 eacc[2] = {};

  for (int k0 = 0; k0 < HIDN; k0 += 32) {
#pragma unroll
    for (int i = 0; i < 2; i++) {
      int lin = tid + i * 256;
      int t   = lin >> 3;
      int col = (lin & 7) << 2;
      size_t row = (size_t)((t * HEADS + hd) * BATCH + b);   // s*B + b
      float4 qv = *reinterpret_cast<const float4*>(Qg + row * HIDN + k0 + col);
      float4 kv = *reinterpret_cast<const float4*>(Kg + row * HIDN + k0 + col);
      __bf16* dq = &Qs[t * 32 + col];
      __bf16* dk = &Ks[t * 32 + col];
      dq[0]=f2bf(qv.x); dq[1]=f2bf(qv.y); dq[2]=f2bf(qv.z); dq[3]=f2bf(qv.w);
      dk[0]=f2bf(kv.x); dk[1]=f2bf(kv.y); dk[2]=f2bf(kv.z); dk[3]=f2bf(kv.w);
    }
    __syncthreads();
    int mq = tm1 * 16 + l16;
    bf16x16 aq = frag16(&Qs[mq * 32 + half * 8], &Qs[mq * 32 + 16 + half * 8]);
#pragma unroll
    for (int j = 0; j < 2; j++) {
      int n = (tn1 + j) * 16 + l16;
      bf16x16 bk = frag16(&Ks[n * 32 + half * 8], &Ks[n * 32 + 16 + half * 8]);
      eacc[j] = __builtin_amdgcn_wmma_f32_16x16x32_bf16(
          false, aq, false, bk, (short)0, eacc[j], false, false);
    }
    __syncthreads();
  }
  const float scale = 0.08838834764831845f;   // 1/sqrt(128)
#pragma unroll
  for (int j = 0; j < 2; j++)
#pragma unroll
    for (int r = 0; r < 8; r++)
      Es[(tm1 * 16 + r + half * 8) * 64 + (tn1 + j) * 16 + l16] = eacc[j][r] * scale;
  __syncthreads();

  // ---------------- softmax (64 rows, one per thread) ----------------
  if (tid < 64) {
    float mx = -1e30f;
    for (int c = 0; c < 64; c++) mx = fmaxf(mx, Es[tid * 64 + c]);
    float sum = 0.f;
    for (int c = 0; c < 64; c++) { float e = __expf(Es[tid * 64 + c] - mx); Es[tid * 64 + c] = e; sum += e; }
    float inv = 1.f / sum;
    for (int c = 0; c < 64; c++) Ab[tid * 64 + c] = f2bf(Es[tid * 64 + c] * inv);
  }
  __syncthreads();

  // ---------------- phase 2: O = att @ V ----------------
  const int tm2 = wid & 3;
  const int ng  = (wid >> 2) * 4;
  for (int n0b = 0; n0b < HIDN; n0b += 128) {
#pragma unroll
    for (int i = 0; i < 8; i++) {
      int lin = tid + i * 256;
      int kk  = lin >> 5;
      int n4  = (lin & 31) << 2;
      size_t row = (size_t)((kk * HEADS + hd) * BATCH + b);
      float4 vv = *reinterpret_cast<const float4*>(Vg + row * HIDN + n0b + n4);
      Vt[(n4 + 0) * 64 + kk] = f2bf(vv.x);
      Vt[(n4 + 1) * 64 + kk] = f2bf(vv.y);
      Vt[(n4 + 2) * 64 + kk] = f2bf(vv.z);
      Vt[(n4 + 3) * 64 + kk] = f2bf(vv.w);
    }
    __syncthreads();

    f32x8 oacc[4] = {};
#pragma unroll
    for (int kc = 0; kc < 64; kc += 32) {
      int ma = tm2 * 16 + l16;
      bf16x16 aa = frag16(&Ab[ma * 64 + kc + half * 8], &Ab[ma * 64 + kc + 16 + half * 8]);
#pragma unroll
      for (int j = 0; j < 4; j++) {
        int n = (ng + j) * 16 + l16;
        bf16x16 bv = frag16(&Vt[n * 64 + kc + half * 8], &Vt[n * 64 + kc + 16 + half * 8]);
        oacc[j] = __builtin_amdgcn_wmma_f32_16x16x32_bf16(
            false, aa, false, bv, (short)0, oacc[j], false, false);
      }
    }
#pragma unroll
    for (int j = 0; j < 4; j++)
#pragma unroll
      for (int r = 0; r < 8; r++) {
        int t = tm2 * 16 + r + half * 8;
        size_t R = (size_t)(t * (BATCH * HEADS) + b * HEADS + hd);
        Og[R * HIDN + n0b + (ng + j) * 16 + l16] = oacc[j][r];
      }
    __syncthreads();
  }
}

// ---------------------------------------------------------------------------
// Full-tensor LayerNorm (deterministic two-stage) + elementwise normalize.
// ---------------------------------------------------------------------------
__global__ __launch_bounds__(256)
void add_reduce_kernel(const float* __restrict__ p, const float* __restrict__ q,
                       float* __restrict__ mid, float* __restrict__ part)
{
  __shared__ float rs[256], rs2[256];
  int tid = threadIdx.x;
  size_t base = (size_t)blockIdx.x * 8192;
  float s = 0.f, s2 = 0.f;
#pragma unroll 8
  for (int i = 0; i < 32; i++) {
    size_t idx = base + (size_t)i * 256 + tid;
    float e = p[idx] + q[idx];
    mid[idx] = e;
    s += e; s2 += e * e;
  }
  rs[tid] = s; rs2[tid] = s2; __syncthreads();
  for (int o = 128; o > 0; o >>= 1) {
    if (tid < o) { rs[tid] += rs[tid + o]; rs2[tid] += rs2[tid + o]; }
    __syncthreads();
  }
  if (tid == 0) { part[blockIdx.x * 2] = rs[0]; part[blockIdx.x * 2 + 1] = rs2[0]; }
}

__global__ __launch_bounds__(256)
void finalize_stats_kernel(const float* __restrict__ part, float* __restrict__ stats)
{
  __shared__ float rs[256], rs2[256];
  int tid = threadIdx.x;
  float s = 0.f, s2 = 0.f;
  for (int i = tid; i < 2048; i += 256) { s += part[i * 2]; s2 += part[i * 2 + 1]; }
  rs[tid] = s; rs2[tid] = s2; __syncthreads();
  for (int o = 128; o > 0; o >>= 1) {
    if (tid < o) { rs[tid] += rs[tid + o]; rs2[tid] += rs2[tid + o]; }
    __syncthreads();
  }
  if (tid == 0) {
    float n = (float)NACT;
    float mean = rs[0] / n;
    float var  = rs2[0] / n - mean * mean;
    stats[0] = mean;
    stats[1] = rsqrtf(var + EPS);
  }
}

__global__ __launch_bounds__(256)
void normalize_kernel(const float* __restrict__ mid, const float* __restrict__ stats,
                      float* __restrict__ out)
{
  float mean = stats[0], inv = stats[1];
  size_t i = ((size_t)blockIdx.x * 256 + threadIdx.x) * 4;
  float4 v = *reinterpret_cast<const float4*>(mid + i);
  float4 o;
  o.x = (v.x - mean) * inv; o.y = (v.y - mean) * inv;
  o.z = (v.z - mean) * inv; o.w = (v.w - mean) * inv;
  *reinterpret_cast<float4*>(out + i) = o;
}

// ---------------------------------------------------------------------------
// Embedding: h[s,b,:] = emb[x[s,b]] * sqrt(H) + pos_emb[s]
// ---------------------------------------------------------------------------
__global__ __launch_bounds__(256)
void embed_kernel(const int* __restrict__ x, const float* __restrict__ emb,
                  const float* __restrict__ pos, float* __restrict__ out)
{
  size_t i = ((size_t)blockIdx.x * 256 + threadIdx.x) * 4;
  size_t row = i >> 10;
  int c = (int)(i & 1023);
  int s = (int)(row >> 5);
  int tok = x[row];
  float4 e = *reinterpret_cast<const float4*>(emb + (size_t)tok * HIDN + c);
  float4 p = *reinterpret_cast<const float4*>(pos + (size_t)s * HIDN + c);
  float4 o;
  o.x = e.x * 32.f + p.x; o.y = e.y * 32.f + p.y;
  o.z = e.z * 32.f + p.z; o.w = e.w * 32.f + p.w;
  *reinterpret_cast<float4*>(out + i) = o;
}

// ---------------------------------------------------------------------------
extern "C" void kernel_launch(void* const* d_in, const int* in_sizes, int n_in,
                              void* d_out, int out_size, void* d_ws, size_t ws_size,
                              hipStream_t stream)
{
  (void)in_sizes; (void)n_in; (void)out_size; (void)ws_size;
  const int*   x   = (const int*)  d_in[0];
  const float* emb = (const float*)d_in[1];
  const float* pos = (const float*)d_in[2];
  const float* wq  = (const float*)d_in[3];
  const float* bq  = (const float*)d_in[4];
  const float* wk  = (const float*)d_in[5];
  const float* bk  = (const float*)d_in[6];
  const float* wv  = (const float*)d_in[7];
  const float* bv  = (const float*)d_in[8];
  const float* wo  = (const float*)d_in[9];
  const float* bo  = (const float*)d_in[10];
  const float* w1  = (const float*)d_in[11];
  const float* b1  = (const float*)d_in[12];
  const float* w2  = (const float*)d_in[13];
  const float* b2  = (const float*)d_in[14];

  float* ws   = (float*)d_ws;
  float* H    = ws;                 // [NACT]
  float* Q    = ws + 1 * NACT;
  float* Kb   = ws + 2 * NACT;
  float* V    = ws + 3 * NACT;
  float* VA   = ws + 4 * NACT;
  float* FFH  = ws + 5 * NACT;      // [4*NACT]
  __bf16* WT  = (__bf16*)(ws + 9 * NACT);           // bf16 transposed weights
  const size_t perLayerWT = 4 * (size_t)HIDN * HIDN + 2 * (size_t)HIDN * FFDIM; // 12M bf16
  float* PART = ws + 9 * NACT + (perLayerWT * NX) / 2 + 64;  // after WT region
  float* STAT = PART + 4096;
  float* MID  = (float*)d_out;

  dim3 blk(256);
  const int nrmGrid = (int)(NACT / 1024);

  // ---- one-shot weight convert+transpose into bf16 [N][K] ----
  for (int i = 0; i < NX; i++) {
    __bf16* base = WT + (size_t)i * perLayerWT;
    __bf16* wtq = base;
    __bf16* wtk = wtq + (size_t)HIDN * HIDN;
    __bf16* wtv = wtk + (size_t)HIDN * HIDN;
    __bf16* wto = wtv + (size_t)HIDN * HIDN;
    __bf16* wt1 = wto + (size_t)HIDN * HIDN;          // [FFDIM][HIDN]
    __bf16* wt2 = wt1 + (size_t)HIDN * FFDIM;         // [HIDN][FFDIM]
    dim3 gs(HIDN / 64, HIDN / 64);
    wconv_kernel<<<gs, blk, 0, stream>>>(wq + (size_t)i * HIDN * HIDN, wtq, HIDN, HIDN);
    wconv_kernel<<<gs, blk, 0, stream>>>(wk + (size_t)i * HIDN * HIDN, wtk, HIDN, HIDN);
    wconv_kernel<<<gs, blk, 0, stream>>>(wv + (size_t)i * HIDN * HIDN, wtv, HIDN, HIDN);
    wconv_kernel<<<gs, blk, 0, stream>>>(wo + (size_t)i * HIDN * HIDN, wto, HIDN, HIDN);
    wconv_kernel<<<dim3(FFDIM / 64, HIDN / 64), blk, 0, stream>>>(
        w1 + (size_t)i * HIDN * FFDIM, wt1, HIDN, FFDIM);
    wconv_kernel<<<dim3(HIDN / 64, FFDIM / 64), blk, 0, stream>>>(
        w2 + (size_t)i * FFDIM * HIDN, wt2, FFDIM, HIDN);
  }

  embed_kernel<<<nrmGrid, blk, 0, stream>>>(x, emb, pos, H);

  for (int i = 0; i < NX; i++) {
    __bf16* base = WT + (size_t)i * perLayerWT;
    const __bf16* wtq = base;
    const __bf16* wtk = wtq + (size_t)HIDN * HIDN;
    const __bf16* wtv = wtk + (size_t)HIDN * HIDN;
    const __bf16* wto = wtv + (size_t)HIDN * HIDN;
    const __bf16* wt1 = wto + (size_t)HIDN * HIDN;
    const __bf16* wt2 = wt1 + (size_t)HIDN * FFDIM;
    const float* bqi = bq + (size_t)i * HIDN;
    const float* bki = bk + (size_t)i * HIDN;
    const float* bvi = bv + (size_t)i * HIDN;
    const float* boi = bo + (size_t)i * HIDN;
    const float* b1i = b1 + (size_t)i * FFDIM;
    const float* b2i = b2 + (size_t)i * HIDN;

    dim3 g1(NTOK / 128, HIDN / 128);    // 128 x 8
    dim3 g2(NTOK / 128, FFDIM / 128);   // 128 x 32

    gemm_bias_kernel<<<g1, blk, 0, stream>>>(H, wtq, bqi, Q,  NTOK, HIDN, HIDN, 0);
    gemm_bias_kernel<<<g1, blk, 0, stream>>>(H, wtk, bki, Kb, NTOK, HIDN, HIDN, 0);
    gemm_bias_kernel<<<g1, blk, 0, stream>>>(H, wtv, bvi, V,  NTOK, HIDN, HIDN, 0);

    attention_kernel<<<BATCH * HEADS, blk, 0, stream>>>(Q, Kb, V, VA);

    gemm_bias_kernel<<<g1, blk, 0, stream>>>(VA, wto, boi, Kb, NTOK, HIDN, HIDN, 0);

    add_reduce_kernel<<<2048, blk, 0, stream>>>(Kb, H, MID, PART);
    finalize_stats_kernel<<<1, blk, 0, stream>>>(PART, STAT);
    normalize_kernel<<<nrmGrid, blk, 0, stream>>>(MID, STAT, MID);

    gemm_bias_kernel<<<g2, blk, 0, stream>>>(MID, wt1, b1i, FFH, NTOK, HIDN, FFDIM, 1);
    gemm_bias_kernel<<<g1, blk, 0, stream>>>(FFH, wt2, b2i, Q,  NTOK, FFDIM, HIDN, 0);

    add_reduce_kernel<<<2048, blk, 0, stream>>>(MID, Q, VA, PART);
    finalize_stats_kernel<<<1, blk, 0, stream>>>(PART, STAT);
    normalize_kernel<<<nrmGrid, blk, 0, stream>>>(VA, STAT,
        (i == NX - 1) ? (float*)d_out : H);
  }
}
